// DeepSurvLoss_61873298866766
// MI455X (gfx1250) — matrix operand order repaired and checked by reading, not compiled, not run on hardware
//
#include <hip/hip_runtime.h>
#include <hip/hip_bf16.h>

typedef __attribute__((ext_vector_type(16))) _Float16 v16h;
typedef __attribute__((ext_vector_type(8)))  float    v8f;

#define NN   16384
#define EPSF 1e-6f

// ---------------- Phase 0: exp(P_risk) (f32 + f16) and per-block max(T) ----
__global__ void cox_k0_exp_tmax(const float* __restrict__ Pr,
                                const float* __restrict__ Tg,
                                float* __restrict__ exp32,
                                _Float16* __restrict__ exp16,
                                float* __restrict__ blockTmax) {
    const int i = blockIdx.x * 256 + threadIdx.x;
    const float e32 = expf(Pr[i]);
    exp32[i] = e32;
    exp16[i] = (_Float16)e32;

    __shared__ float sb[256];
    sb[threadIdx.x] = Tg[i];
    __syncthreads();
    #pragma unroll
    for (int s = 128; s > 0; s >>= 1) {
        if (threadIdx.x < s) sb[threadIdx.x] = fmaxf(sb[threadIdx.x], sb[threadIdx.x + s]);
        __syncthreads();
    }
    if (threadIdx.x == 0) blockTmax[blockIdx.x] = sb[0];
}

// ---------------- Phase 0b: reduce 64 block maxima -> Tmax -----------------
__global__ void cox_k0b_tmax(const float* __restrict__ blockTmax,
                             float* __restrict__ Tmax) {
    __shared__ float sb[64];
    sb[threadIdx.x] = blockTmax[threadIdx.x];
    __syncthreads();
    #pragma unroll
    for (int s = 32; s > 0; s >>= 1) {
        if (threadIdx.x < s) sb[threadIdx.x] = fmaxf(sb[threadIdx.x], sb[threadIdx.x + s]);
        __syncthreads();
    }
    if (threadIdx.x == 0) Tmax[0] = sb[0];
}

// ---------------- Phase 1: WMMA risk-set sums ------------------------------
// P_exp_sum[i] = sum_j (T[i] < T[j]) * exp16[j]
// 8 waves/block, 16 i-rows per wave => 128 rows/block, 128 blocks = 16384.
// j is staged through LDS in chunks of 1024 (T: 4KB f32, exp: 2KB f16).
#define JCHUNK 1024
__global__ void cox_k1_wmma_risksum(const float* __restrict__ Tg,
                                    const _Float16* __restrict__ exp16,
                                    float* __restrict__ psum) {
    __shared__ float    sT[JCHUNK];
    __shared__ _Float16 sP[JCHUNK];

    const int lane  = threadIdx.x & 31;
    const int wave  = threadIdx.x >> 5;          // 0..7
    const int m     = lane & 15;                 // row within 16-row tile
    const int khalf = (lane >> 4) & 1;           // K-half select (A/B layouts)
    const int ibase = blockIdx.x * 128 + wave * 16;

    const float Tm = Tg[ibase + m];              // this lane's row threshold

    v8f acc = {};                                // f32 16x16 accumulator (8 VGPRs)

    for (int jc = 0; jc < NN; jc += JCHUNK) {
        // prefetch next chunk into caches while we still compute this one
        if (jc + JCHUNK < NN) {
            __builtin_prefetch(&Tg[jc + JCHUNK + ((threadIdx.x & 63) << 4)], 0, 1);
            __builtin_prefetch(&exp16[jc + JCHUNK + ((threadIdx.x & 63) << 4)], 0, 1);
        }
        // cooperative stage of the j-chunk into LDS (uniform trip count)
        #pragma unroll
        for (int t = 0; t < JCHUNK / 256; ++t) {
            const int idx = threadIdx.x + t * 256;
            sT[idx] = Tg[jc + idx];
            sP[idx] = exp16[jc + idx];
        }
        __syncthreads();

        for (int jj = 0; jj < JCHUNK; jj += 32) {
            v16h a, b;
            // A: 16x32 f16 mask tile, ISA layout:
            //   lanes m=0..15; khalf=0 -> K in {0..7,16..23}, khalf=1 -> {8..15,24..31}
            // B: 32x16 f16, broadcast across N columns:
            //   same K order as A: khalf=0 -> K in {0..7,16..23}, khalf=1 -> {8..15,24..31}
            #pragma unroll
            for (int e = 0; e < 16; ++e) {
                const int v = e >> 1, s = e & 1;
                const int k = ((v < 4) ? (2 * v + s) : (16 + 2 * (v - 4) + s)) + khalf * 8;
                a[e] = (Tm < sT[jj + k]) ? (_Float16)1.0f : (_Float16)0.0f;
                b[e] = sP[jj + k];
            }
            acc = __builtin_amdgcn_wmma_f32_16x16x32_f16(
                      /*neg_a=*/false, a, /*neg_b=*/false, b,
                      /*c_mod=*/(short)0, acc, /*reuse_a=*/false, /*reuse_b=*/false);
        }
        __syncthreads();
    }

    // D layout: lane (N=lane&15), VGPR r holds M = r + 8*khalf.
    // All N columns identical (B broadcast) -> lanes with N==0 write.
    if ((lane & 15) == 0) {
        const int ib = ibase + khalf * 8;
        #pragma unroll
        for (int r = 0; r < 8; ++r) psum[ib + r] = acc[r];
    }
}

// ---------------- Phase 2: per-element terms + per-block partial sums ------
__global__ void cox_k2_terms(const float* __restrict__ Tg,
                             const int* __restrict__ Eg,
                             const float* __restrict__ exp32,
                             const float* __restrict__ psum,
                             const float* __restrict__ Tmax,
                             float* __restrict__ pnum,
                             float* __restrict__ pden) {
    const int i = blockIdx.x * 256 + threadIdx.x;
    const float tmax = Tmax[0];
    const float ef = ((Eg[i] != 0) && (Tg[i] < tmax)) ? 1.0f : 0.0f;
    const float pt = exp32[i] / (psum[i] + EPSF);
    const float pc = fmaxf(pt, EPSF);           // upper clip is a value no-op
    float numv = ef * logf(pc);

    __shared__ float sb[256];
    sb[threadIdx.x] = numv;
    __syncthreads();
    #pragma unroll
    for (int s = 128; s > 0; s >>= 1) {
        if (threadIdx.x < s) sb[threadIdx.x] += sb[threadIdx.x + s];
        __syncthreads();
    }
    if (threadIdx.x == 0) pnum[blockIdx.x] = sb[0];
    __syncthreads();

    sb[threadIdx.x] = ef;
    __syncthreads();
    #pragma unroll
    for (int s = 128; s > 0; s >>= 1) {
        if (threadIdx.x < s) sb[threadIdx.x] += sb[threadIdx.x + s];
        __syncthreads();
    }
    if (threadIdx.x == 0) pden[blockIdx.x] = sb[0];
}

// ---------------- Phase 3: final deterministic reduction -------------------
__global__ void cox_k3_final(const float* __restrict__ pnum,
                             const float* __restrict__ pden,
                             float* __restrict__ out) {
    __shared__ float sn[64];
    __shared__ float sd[64];
    sn[threadIdx.x] = pnum[threadIdx.x];
    sd[threadIdx.x] = pden[threadIdx.x];
    __syncthreads();
    #pragma unroll
    for (int s = 32; s > 0; s >>= 1) {
        if (threadIdx.x < s) {
            sn[threadIdx.x] += sn[threadIdx.x + s];
            sd[threadIdx.x] += sd[threadIdx.x + s];
        }
        __syncthreads();
    }
    if (threadIdx.x == 0) out[0] = -sn[0] / sd[0];
}

extern "C" void kernel_launch(void* const* d_in, const int* in_sizes, int n_in,
                              void* d_out, int out_size, void* d_ws, size_t ws_size,
                              hipStream_t stream) {
    const float* Pr = (const float*)d_in[0];   // P_risk f32 [N]
    const float* Tg = (const float*)d_in[1];   // T      f32 [N]
    const int*   Eg = (const int*)d_in[2];     // E      i32 [N]
    float* out = (float*)d_out;

    char* ws = (char*)d_ws;
    float*    exp32     = (float*)(ws);                 // 65536 B
    float*    psum      = (float*)(ws + 65536);         // 65536 B
    _Float16* exp16     = (_Float16*)(ws + 131072);     // 32768 B
    float*    blockTmax = (float*)(ws + 163840);        //   256 B
    float*    Tmax      = (float*)(ws + 164096);        //     4 B
    float*    pnum      = (float*)(ws + 164352);        //   256 B
    float*    pden      = (float*)(ws + 164608);        //   256 B

    cox_k0_exp_tmax<<<NN / 256, 256, 0, stream>>>(Pr, Tg, exp32, exp16, blockTmax);
    cox_k0b_tmax<<<1, 64, 0, stream>>>(blockTmax, Tmax);
    cox_k1_wmma_risksum<<<NN / 128, 256, 0, stream>>>(Tg, exp16, psum);
    cox_k2_terms<<<NN / 256, 256, 0, stream>>>(Tg, Eg, exp32, psum, Tmax, pnum, pden);
    cox_k3_final<<<1, 64, 0, stream>>>(pnum, pden, out);
}
